// multihead_2327872274483
// MI455X (gfx1250) — matrix operand order repaired and checked by reading, not compiled, not run on hardware
//
#include <hip/hip_runtime.h>
#include <hip/hip_bf16.h>

// ---------------------------------------------------------------------------
// Multi-head causal attention, bf16 WMMA pipeline for gfx1250 (MI455X).
//   K1: QKV projection -> Q[T][64], K[T][64], V^T[48][T]   (bf16, zero-padded)
//   K2: per-(b,h): S = QK^T*scale (causal), softmax, O = P V -> Y concat (bf16)
//   K3: out = Y @ Wo^T + bo  (f32)
// All fragments are contiguous 16-byte LDS/global accesses (2x b128 each).
// B fragments live in distinct-register arrays (batched loads, graduated
// waits) and each feeds both M-tile accumulators.
// ---------------------------------------------------------------------------

#define EMBEDS    195
#define BLOCK_T   256
#define HEADS     5
#define HEAD_SIZE 39
#define BATCH     128

#define CPAD   224   // embeds padded to 7 K-steps of 32
#define DPAD   48    // head dim padded to 3 N-tiles of 16
#define DPAD2  64    // head dim padded to 2 K-steps of 32 (score GEMM K-dim)
#define NPAD   208   // output cols padded to 13 N-tiles of 16
#define SSTR   260   // score f32 row stride: rows 16B-aligned (260*4=1040=65*16)

typedef __bf16 bf16;
typedef __attribute__((ext_vector_type(16))) __bf16 v16bf;
typedef __attribute__((ext_vector_type(8)))  __bf16 v8bf;
typedef __attribute__((ext_vector_type(8)))  float  v8f;
typedef __attribute__((ext_vector_type(4)))  float  v4f;

__device__ __forceinline__ v8f wmma_bf16(v16bf a, v16bf b, v8f c) {
  // D = A(16x32 bf16) * B(32x16 bf16) + C(16x16 f32)
  return __builtin_amdgcn_wmma_f32_16x16x32_bf16(false, a, false, b, (short)0, c,
                                                 false, false);
}
__device__ __forceinline__ v16bf cat16(v8bf lo, v8bf hi) {
  return __builtin_shufflevector(lo, hi, 0, 1, 2, 3, 4, 5, 6, 7,
                                         8, 9, 10, 11, 12, 13, 14, 15);
}
// A fragment (row-major MxK tile): lane reads K = half+0..7 and half+16..23.
__device__ __forceinline__ v16bf load_afrag(const bf16* rowk, int half) {
  return cat16(*(const v8bf*)(rowk + half), *(const v8bf*)(rowk + half + 16));
}
// B fragment (N-major KxN tile, i.e. B^T row-major): K = bhalf+0..7 and bhalf+16..23.
__device__ __forceinline__ v16bf load_bfrag(const bf16* p) {
  return cat16(*(const v8bf*)p, *(const v8bf*)(p + 16));
}

// ---------------------------------------------------------------------------
// Kernel 1: QKV projection. grid = B*H, 256 threads (8 waves).
// ---------------------------------------------------------------------------
__global__ __launch_bounds__(256) void qkv_kernel(
    const float* __restrict__ x,    // [B,T,C]
    const float* __restrict__ wq,   // [H,C,d]
    const float* __restrict__ wk,
    const float* __restrict__ wv,
    bf16* __restrict__ Q,           // [B*H, T, DPAD2]  (cols 48..63 zeroed)
    bf16* __restrict__ K,           // [B*H, T, DPAD2]  (cols 48..63 zeroed)
    bf16* __restrict__ Vt)          // [B*H, DPAD, T]   (V transposed)
{
  extern __shared__ char smem[];
  bf16* XL  = (bf16*)smem;                        // [256][CPAD]       zero-padded
  bf16* WLT = (bf16*)(smem + BLOCK_T * CPAD * 2); // [3][DPAD][CPAD]   W^T, zero-padded

  const int tid  = threadIdx.x;
  const int lane = tid & 31;
  const int wave = tid >> 5;
  const int bh   = blockIdx.x;
  const int b    = bh / HEADS;
  const int h    = bh % HEADS;

  bf16* Qb  = Q  + (size_t)bh * BLOCK_T * DPAD2;
  bf16* Kb  = K  + (size_t)bh * BLOCK_T * DPAD2;
  bf16* Vtb = Vt + (size_t)bh * DPAD * BLOCK_T;

  // Zero the d-padding tails of Q and K (cols 48..63) with 16B stores.
  {
    const v8bf z8 = {};
    for (int t = tid; t < BLOCK_T; t += 256) {
      *(v8bf*)&Qb[t * DPAD2 + 48] = z8; *(v8bf*)&Qb[t * DPAD2 + 56] = z8;
      *(v8bf*)&Kb[t * DPAD2 + 48] = z8; *(v8bf*)&Kb[t * DPAD2 + 56] = z8;
    }
  }

  // Stage X[b] row-per-thread (256 rows, 256 threads), zero-pad K to CPAD.
  {
    const float* xr = x + ((size_t)b * BLOCK_T + tid) * EMBEDS;
    bf16* xl = XL + tid * CPAD;
    for (int c = 0; c < EMBEDS; ++c) xl[c] = (bf16)xr[c];
    for (int c = EMBEDS; c < CPAD; ++c) xl[c] = (bf16)0.0f;
  }
  // Stage W^T rows: 3*DPAD = 144 rows of CPAD, row (mat,dd) per thread.
  for (int rI = tid; rI < 3 * DPAD; rI += 256) {
    int m = rI / DPAD, dd = rI % DPAD;
    const float* wp = ((m == 0) ? wq : (m == 1) ? wk : wv) + (size_t)h * EMBEDS * HEAD_SIZE;
    bf16* wl = WLT + rI * CPAD;
    if (dd < HEAD_SIZE)
      for (int c = 0; c < EMBEDS; ++c) wl[c] = (bf16)wp[c * HEAD_SIZE + dd];
    else
      for (int c = 0; c < EMBEDS; ++c) wl[c] = (bf16)0.0f;
    for (int c = EMBEDS; c < CPAD; ++c) wl[c] = (bf16)0.0f;
  }
  __syncthreads();

  const int half  = (lane < 16) ? 0 : 8;   // A-fragment K offset
  const int bhalf = (lane < 16) ? 0 : 8;   // B-fragment K offset
  const int mrow  = lane & 15;

  // Cache A fragments for BOTH M-tiles of this wave (shared by q/k/v).
  v16bf afr[2][7];
#pragma unroll
  for (int mt = 0; mt < 2; ++mt) {
    const bf16* xrow = XL + ((wave * 2 + mt) * 16 + mrow) * CPAD;
#pragma unroll
    for (int kk = 0; kk < 7; ++kk) afr[mt][kk] = load_afrag(xrow + kk * 32, half);
  }

#pragma unroll
  for (int mat = 0; mat < 3; ++mat) {     // unrolled -> direct store targets
    for (int nt = 0; nt < 3; ++nt) {
      const bf16* wrow = WLT + (mat * DPAD + nt * 16 + mrow) * CPAD + bhalf;
      v16bf bfr[7];
#pragma unroll
      for (int kk = 0; kk < 7; ++kk) bfr[kk] = load_bfrag(wrow + kk * 32);
#pragma unroll
      for (int mt = 0; mt < 2; ++mt) {    // B frags feed both M-tiles
        v8f acc = {};
#pragma unroll
        for (int kk = 0; kk < 7; ++kk) acc = wmma_bf16(afr[mt][kk], bfr[kk], acc);
        const int tbase = (wave * 2 + mt) * 16;
        const int dd = nt * 16 + mrow;
#pragma unroll
        for (int r = 0; r < 8; ++r) {
          int t = tbase + ((lane < 16) ? r : r + 8);
          bf16 v = (bf16)acc[r];
          if (mat == 0)      Qb [t * DPAD2 + dd] = v;
          else if (mat == 1) Kb [t * DPAD2 + dd] = v;
          else               Vtb[dd * BLOCK_T + t] = v;
        }
      }
    }
  }
}

// ---------------------------------------------------------------------------
// Kernel 2: attention per (b,h). grid = B*H, 256 threads (8 waves).
// Each wave owns 32 query rows end-to-end: no barriers after the K/V stage.
// ---------------------------------------------------------------------------
__global__ __launch_bounds__(256) void attn_kernel(
    const bf16* __restrict__ Q,    // [B*H, T, DPAD2]
    const bf16* __restrict__ K,    // [B*H, T, DPAD2]
    const bf16* __restrict__ Vt,   // [B*H, DPAD, T]
    bf16* __restrict__ Y)          // [B, T, EMBEDS] (concat heads)
{
  extern __shared__ char smem[];
  float* S  = (float*)smem;                                  // [256][SSTR] row-major
  bf16*  KL = (bf16*)(smem + (size_t)BLOCK_T * SSTR * 4);    // [T][DPAD2]
  bf16*  VL = (bf16*)(smem + (size_t)BLOCK_T * SSTR * 4
                           + (size_t)BLOCK_T * DPAD2 * 2);   // [DPAD][T]  (V^T)

  const int tid  = threadIdx.x;
  const int lane = tid & 31;
  const int wave = tid >> 5;
  const int bh   = blockIdx.x;
  const float scale = 0.07161148740394329f;  // 1/sqrt(EMBEDS)

  { // stage K and V^T into LDS with 16B vector copies
    const bf16* kg = K + (size_t)bh * BLOCK_T * DPAD2;
    for (int i = tid * 8; i < BLOCK_T * DPAD2; i += 256 * 8)
      *(v8bf*)&KL[i] = *(const v8bf*)&kg[i];
    const bf16* vg = Vt + (size_t)bh * DPAD * BLOCK_T;
    for (int i = tid * 8; i < DPAD * BLOCK_T; i += 256 * 8)
      *(v8bf*)&VL[i] = *(const v8bf*)&vg[i];
  }
  __syncthreads();

  const int m0    = wave * 32;
  const int half  = (lane < 16) ? 0 : 8;
  const int bhalf = (lane < 16) ? 0 : 8;
  const int mrow  = lane & 15;

  // Q fragments for this wave's 32 rows (2 M-tiles x 2 K-steps) from global.
  v16bf aq[2][2];
  {
    const bf16* qg = Q + (size_t)bh * BLOCK_T * DPAD2;
#pragma unroll
    for (int mt = 0; mt < 2; ++mt) {
      const bf16* qrow = qg + (size_t)(m0 + mt * 16 + mrow) * DPAD2;
#pragma unroll
      for (int ks = 0; ks < 2; ++ks) aq[mt][ks] = load_afrag(qrow + ks * 32, half);
    }
  }

  // Scores: S[row][col] = scale * q[row].k[col], causal-masked.
  // K fragments loaded once per N-tile, feed both M-tile accumulators.
  for (int nt = 0; nt < 16; ++nt) {
    const bf16* krow = KL + (nt * 16 + mrow) * DPAD2 + bhalf;
    const v16bf bk0 = load_bfrag(krow);
    const v16bf bk1 = load_bfrag(krow + 32);
    const int col = nt * 16 + mrow;
#pragma unroll
    for (int mt = 0; mt < 2; ++mt) {
      v8f acc = {};
      acc = wmma_bf16(aq[mt][0], bk0, acc);
      acc = wmma_bf16(aq[mt][1], bk1, acc);
#pragma unroll
      for (int r = 0; r < 8; ++r) {
        int row = m0 + mt * 16 + ((lane < 16) ? r : r + 8);
        float v = acc[r] * scale;
        if (col > row) v = -3.4028235e38f;  // causal mask (finfo.min)
        S[row * SSTR + col] = v;
      }
    }
  }

  // Row softmax: lane L owns row m0+L (wave-private rows -> no barrier).
  {
    float* Sr = S + (m0 + lane) * SSTR;
    float mx = -3.4028235e38f;
    for (int c = 0; c < BLOCK_T; c += 4) {
      v4f v = *(const v4f*)&Sr[c];
      mx = fmaxf(mx, fmaxf(fmaxf(v[0], v[1]), fmaxf(v[2], v[3])));
    }
    float sum = 0.0f;
    for (int c = 0; c < BLOCK_T; c += 4) {
      v4f v = *(const v4f*)&Sr[c];
      v4f e;
#pragma unroll
      for (int j = 0; j < 4; ++j) { e[j] = __expf(v[j] - mx); sum += e[j]; }
      *(v4f*)&Sr[c] = e;
    }
    float inv = 1.0f / sum;
    for (int c = 0; c < BLOCK_T; c += 4) {
      v4f v = *(const v4f*)&Sr[c];
#pragma unroll
      for (int j = 0; j < 4; ++j) v[j] *= inv;
      *(v4f*)&Sr[c] = v;
    }
  }

  // O = P V: preload ALL P fragments (one f32->bf16 cvt pass), then per
  // N-tile preload ALL V fragments (distinct regs -> batched ds_load_b128),
  // each feeding both M-tile accumulators.
  v16bf ap[2][8];
#pragma unroll
  for (int mt = 0; mt < 2; ++mt) {
    const float* Srow = S + (m0 + mt * 16 + mrow) * SSTR;
#pragma unroll
    for (int ks = 0; ks < 8; ++ks) {
      const float* Sp = Srow + ks * 32 + half;
      v4f p0 = *(const v4f*)(Sp);
      v4f p1 = *(const v4f*)(Sp + 4);
      v4f p2 = *(const v4f*)(Sp + 16);
      v4f p3 = *(const v4f*)(Sp + 20);
      v16bf a;
#pragma unroll
      for (int j = 0; j < 4; ++j) {
        a[j]      = (bf16)p0[j];
        a[4 + j]  = (bf16)p1[j];
        a[8 + j]  = (bf16)p2[j];
        a[12 + j] = (bf16)p3[j];
      }
      ap[mt][ks] = a;
    }
  }

  const int b = bh / HEADS, h = bh % HEADS;
  for (int nt = 0; nt < 3; ++nt) {
    const bf16* vrow = VL + (nt * 16 + mrow) * BLOCK_T + bhalf;
    v16bf bv[8];
#pragma unroll
    for (int ks = 0; ks < 8; ++ks) bv[ks] = load_bfrag(vrow + ks * 32);
    v8f acc0 = {}, acc1 = {};
#pragma unroll
    for (int ks = 0; ks < 8; ++ks) {
      acc0 = wmma_bf16(ap[0][ks], bv[ks], acc0);
      acc1 = wmma_bf16(ap[1][ks], bv[ks], acc1);
    }
    const int dd = nt * 16 + mrow;
#pragma unroll
    for (int mt = 0; mt < 2; ++mt) {
      const v8f acc = (mt == 0) ? acc0 : acc1;
#pragma unroll
      for (int r = 0; r < 8; ++r) {
        int t = m0 + mt * 16 + ((lane < 16) ? r : r + 8);
        if (dd < HEAD_SIZE)
          Y[((size_t)b * BLOCK_T + t) * EMBEDS + h * HEAD_SIZE + dd] = (bf16)acc[r];
      }
    }
  }
}

// ---------------------------------------------------------------------------
// Kernel 3: output projection. grid = B, 256 threads (8 waves).
// out[b,t,o] = sum_c Y[b,t,c] * Wo[o,c] + bo[o]
// ---------------------------------------------------------------------------
__global__ __launch_bounds__(256) void proj_kernel(
    const bf16*  __restrict__ Y,    // [B,T,EMBEDS]
    const float* __restrict__ wo,   // [EMBEDS, EMBEDS] (out, in) -> already N-major
    const float* __restrict__ bo,   // [EMBEDS]
    float* __restrict__ out)        // [B,T,EMBEDS]
{
  extern __shared__ char smem[];
  bf16* YL  = (bf16*)smem;                         // [256][CPAD]  zero-padded
  bf16* WOL = (bf16*)(smem + BLOCK_T * CPAD * 2);  // [NPAD][CPAD] WOL[o][c], zero-padded

  const int tid  = threadIdx.x;
  const int lane = tid & 31;
  const int wave = tid >> 5;
  const int b    = blockIdx.x;

  { // stage Y[b] row-per-thread
    const bf16* yr = Y + ((size_t)b * BLOCK_T + tid) * EMBEDS;
    bf16* yl = YL + tid * CPAD;
    for (int c = 0; c < EMBEDS; ++c) yl[c] = yr[c];
    for (int c = EMBEDS; c < CPAD; ++c) yl[c] = (bf16)0.0f;
  }
  for (int o = tid; o < NPAD; o += 256) {
    bf16* wl = WOL + o * CPAD;
    if (o < EMBEDS) {
      const float* wr = wo + (size_t)o * EMBEDS;
      for (int c = 0; c < EMBEDS; ++c) wl[c] = (bf16)wr[c];
    } else {
      for (int c = 0; c < EMBEDS; ++c) wl[c] = (bf16)0.0f;
    }
    for (int c = EMBEDS; c < CPAD; ++c) wl[c] = (bf16)0.0f;
  }
  __syncthreads();

  const int half  = (lane < 16) ? 0 : 8;
  const int bhalf = (lane < 16) ? 0 : 8;
  const int mrow  = lane & 15;

  // Cache A fragments for both M-tiles; Wo fragments batch-loaded per N-tile.
  v16bf afr[2][7];
#pragma unroll
  for (int mt = 0; mt < 2; ++mt) {
    const bf16* yrow = YL + ((wave * 2 + mt) * 16 + mrow) * CPAD;
#pragma unroll
    for (int kk = 0; kk < 7; ++kk) afr[mt][kk] = load_afrag(yrow + kk * 32, half);
  }

  for (int nt = 0; nt < 13; ++nt) {
    const bf16* wrow = WOL + (nt * 16 + mrow) * CPAD + bhalf;
    v16bf bw[7];
#pragma unroll
    for (int kk = 0; kk < 7; ++kk) bw[kk] = load_bfrag(wrow + kk * 32);
    v8f acc0 = {}, acc1 = {};
#pragma unroll
    for (int kk = 0; kk < 7; ++kk) {
      acc0 = wmma_bf16(afr[0][kk], bw[kk], acc0);
      acc1 = wmma_bf16(afr[1][kk], bw[kk], acc1);
    }
    const int o = nt * 16 + mrow;
    const float bias = (o < EMBEDS) ? bo[o] : 0.0f;
#pragma unroll
    for (int mt = 0; mt < 2; ++mt) {
      const v8f acc = (mt == 0) ? acc0 : acc1;
      const int tbase = (wave * 2 + mt) * 16;
#pragma unroll
      for (int r = 0; r < 8; ++r) {
        int t = tbase + ((lane < 16) ? r : r + 8);
        if (o < EMBEDS)
          out[((size_t)b * BLOCK_T + t) * EMBEDS + o] = acc[r] + bias;
      }
    }
  }
}

// ---------------------------------------------------------------------------
extern "C" void kernel_launch(void* const* d_in, const int* in_sizes, int n_in,
                              void* d_out, int out_size, void* d_ws, size_t ws_size,
                              hipStream_t stream) {
  const float* x  = (const float*)d_in[0];
  const float* wq = (const float*)d_in[1];
  const float* wk = (const float*)d_in[2];
  const float* wv = (const float*)d_in[3];
  const float* wo = (const float*)d_in[4];
  const float* bo = (const float*)d_in[5];
  float* out = (float*)d_out;

  // Workspace layout (bytes):
  //   Q : B*H*T*64  bf16 = 20,971,520
  //   K : B*H*T*64  bf16 = 20,971,520
  //   Vt: B*H*48*T  bf16 = 15,728,640
  //   Y : B*T*195   bf16 = 12,779,520   (total ~70.5 MB)
  char* ws = (char*)d_ws;
  bf16* Q  = (bf16*)(ws);
  bf16* K  = (bf16*)(ws + 20971520u);
  bf16* Vt = (bf16*)(ws + 41943040u);
  bf16* Y  = (bf16*)(ws + 57671680u);

  const size_t lds1 = (size_t)BLOCK_T * CPAD * 2 + (size_t)3 * DPAD * CPAD * 2;  // 179,200 B
  const size_t lds2 = (size_t)BLOCK_T * SSTR * 4 + (size_t)BLOCK_T * DPAD2 * 2
                    + (size_t)DPAD * BLOCK_T * 2;                                // 323,584 B
  const size_t lds3 = (size_t)BLOCK_T * CPAD * 2 + (size_t)NPAD * CPAD * 2;      // 207,872 B

  qkv_kernel <<<BATCH * HEADS, 256, lds1, stream>>>(x, wq, wk, wv, Q, K, Vt);
  attn_kernel<<<BATCH * HEADS, 256, lds2, stream>>>(Q, K, Vt, Y);
  proj_kernel<<<BATCH,         256, lds3, stream>>>(Y, wo, bo, out);
}
